// MixtureOfExperts_88751204204902
// MI455X (gfx1250) — matrix operand order repaired and checked by reading, not compiled, not run on hardware
//
#include <hip/hip_runtime.h>
#include <hip/hip_bf16.h>

// ---------------- problem dims (match reference) ----------------
#define T_TOK 4096
#define NE    8
#define DM    1024
#define FF    4096
#define CAP   1536
#define MT    32                  // slot rows per workgroup (2 row-groups of 16)
#define NRG   2                   // row groups
#define TILES_PER_E (CAP / MT)    // 48
#define FCHUNK 128                // f columns per pipeline stage (8 waves * 16)

typedef __bf16 bf16_t;
typedef bf16_t v16bf __attribute__((ext_vector_type(16)));
typedef float  v8f   __attribute__((ext_vector_type(8)));

// workspace layout (bytes)
#define W1T_BYTES ((size_t)NE * FF * DM * 2)   // 64 MB bf16  W1t[e][f][d]
#define W2T_BYTES ((size_t)NE * DM * FF * 2)   // 64 MB bf16  W2t[e][d][f]
#define IDX_BYTES ((size_t)NE * CAP * 4)
#define GAIN_BYTES ((size_t)NE * CAP * 4)

__device__ __forceinline__ void atomic_add_f32(float* p, float v) {
  __hip_atomic_fetch_add(p, v, __ATOMIC_RELAXED, __HIP_MEMORY_SCOPE_AGENT);
}

// ---------------- kernel 1: zero output + routing state ----------------
__global__ void moe_init(float* __restrict__ y, int* __restrict__ idx,
                         float* __restrict__ gain, int* __restrict__ cnt) {
  size_t i = (size_t)blockIdx.x * blockDim.x + threadIdx.x;
  size_t stride = (size_t)gridDim.x * blockDim.x;
  for (size_t j = i; j < (size_t)T_TOK * DM; j += stride) y[j] = 0.f;
  for (size_t j = i; j < (size_t)NE * CAP; j += stride) { idx[j] = 0; gain[j] = 0.f; }
  if (i < NE) cnt[i] = 0;
}

// ---------------- kernel 2: routing (slot append per expert) ----------------
// route_weight holds softmax weights (>0) exactly at routed positions, 0 elsewhere,
// so w>0 <=> route_mask. Expected tokens/expert = 1024 < C = 1536, so the
// capacity top-C reduces to "take all routed tokens" (order-independent sum).
__global__ void moe_route(const float* __restrict__ rw, int* __restrict__ idx,
                          float* __restrict__ gain, int* __restrict__ cnt) {
  int t = blockIdx.x * blockDim.x + threadIdx.x;
  if (t >= T_TOK) return;
  #pragma unroll
  for (int e = 0; e < NE; ++e) {
    float w = rw[t * NE + e];
    if (w > 0.f) {
      int s = atomicAdd(&cnt[e], 1);
      if (s < CAP) { idx[e * CAP + s] = t; gain[e * CAP + s] = w; }
    }
  }
}

// ---------------- kernel 3: weight transpose + f32->bf16 ----------------
// W1t[e][f][d] = bf16(W1[e][d][f]) ; W2t[e][d][f] = bf16(W2[e][f][d])
// K-contiguous-per-lane layouts make every WMMA B operand two global b128 loads.
__global__ void moe_prep(const float* __restrict__ W1, const float* __restrict__ W2,
                         bf16_t* __restrict__ W1t, bf16_t* __restrict__ W2t) {
  size_t n = (size_t)NE * FF * DM;
  size_t stride = (size_t)gridDim.x * blockDim.x;
  for (size_t i = (size_t)blockIdx.x * blockDim.x + threadIdx.x; i < n; i += stride) {
    size_t e = i / ((size_t)FF * DM);
    size_t r = i % ((size_t)FF * DM);
    size_t f = r / DM, d = r % DM;
    W1t[i] = (bf16_t)W1[(e * DM + d) * FF + f];                     // write coalesced
    W2t[(e * DM + d) * FF + f] = (bf16_t)W2[(e * FF + f) * DM + d]; // read coalesced
  }
}

// ---------------- kernel 4: fused gather + MLP + combine ----------------
// 1 workgroup = 8 wave32 handles (expert e, 32 slot rows), all 1024 out cols.
// Two 16-row groups share every B operand -> 2 WMMAs per B load (2x L2 reuse).
// hLDS is double-buffered -> exactly one workgroup barrier per f-chunk.
// A-operand LDS layout: lane' = m + 16*((k>>4)&1), elem' = k%16.
__global__ __launch_bounds__(256)
void moe_ffn(const float* __restrict__ x, const float* __restrict__ b1,
             const float* __restrict__ b2, const bf16_t* __restrict__ W1t,
             const bf16_t* __restrict__ W2t, const int* __restrict__ idx,
             const float* __restrict__ gain, float* __restrict__ y) {
  __shared__ __align__(32) bf16_t aLDS[DM / 32][NRG][32][16];         // x tile, 64 KB
  __shared__ __align__(32) bf16_t hLDS[2][FCHUNK / 32][NRG][32][16];  // h, 2x8 KB

  __shared__ float gLDS[MT];
  __shared__ int   tokLDS[MT];

  const int e    = blockIdx.x / TILES_PER_E;
  const int m0   = (blockIdx.x % TILES_PER_E) * MT;
  const int tid  = threadIdx.x;
  const int wave = tid >> 5;
  const int lane = tid & 31;

  if (tid < MT) {
    int s = e * CAP + m0 + tid;
    tokLDS[tid] = idx[s];
    gLDS[tid]   = gain[s];
  }
  __syncthreads();

  // gather x rows -> bf16 -> LDS (A-operand layout); coalesced along k
  for (int i = tid; i < MT * DM; i += 256) {
    int m = i >> 10;            // i / DM : 0..31
    int k = i & (DM - 1);
    float v = (gLDS[m] > 0.f) ? x[(size_t)tokLDS[m] * DM + k] : 0.f;
    int hi = (k >> 4) & 1;
    aLDS[k >> 5][m >> 4][(m & 15) + (hi << 4)]
        [k & 15] = (bf16_t)v;
  }
  __syncthreads();

  v8f oacc[NRG][8] = {};                  // wave's 32 x 128 output accumulator
  const int fl    = lane & 15;            // N index inside a 16-wide tile
  const int khalf = (lane >> 4) << 4;     // lanes 16..31 read K+16 half
  const int mrow  = (lane >> 4) << 3;     // C/D layout row base for this lane

  const bf16_t* w1p0 =
      W1t + ((size_t)e * FF + wave * 16 + fl) * DM + khalf;
  const bf16_t* w2p0 =
      W2t + ((size_t)e * DM + wave * FCHUNK + fl) * FF + khalf;

  for (int fc = 0; fc < FF / FCHUNK; ++fc) {
    const int f0  = fc * FCHUNK;
    const int buf = fc & 1;
    const bf16_t* w1p = w1p0 + (size_t)f0 * DM;
    const bf16_t* w2p = w2p0 + f0;

    // pull next f-chunk's weight rows toward this WGP while we compute
    if (fc + 1 < FF / FCHUNK) {
      __builtin_prefetch(w1p + (size_t)FCHUNK * DM, 0, 0);
      __builtin_prefetch(w2p + FCHUNK, 0, 0);
    }

    // --- GEMM1: this wave's 2 x (16x16) h tiles, K = D = 1024 ---
    v8f hacc[NRG] = {};
    #pragma unroll 4
    for (int kt = 0; kt < DM / 32; ++kt) {
      v16bf b = *(const v16bf*)(w1p + kt * 32);        // shared by both row groups
      #pragma unroll
      for (int rg = 0; rg < NRG; ++rg) {
        v16bf a = *(const v16bf*)&aLDS[kt][rg][lane][0];
        hacc[rg] = __builtin_amdgcn_wmma_f32_16x16x32_bf16(
            false, a, false, b, (short)0, hacc[rg], false, false);
      }
    }
    float bias1 = b1[e * FF + f0 + wave * 16 + fl];

    // write h into buffer `buf`; readers of this buffer from iteration fc-2
    // are already past the barrier of iteration fc-1, so no pre-write barrier.
    #pragma unroll
    for (int rg = 0; rg < NRG; ++rg) {
      #pragma unroll
      for (int j = 0; j < 8; ++j) {
        float hv = hacc[rg][j] + bias1;
        hv = hv > 0.f ? hv : 0.f;                      // relu
        int m  = j + mrow;                             // C/D layout row
        int kk = wave * 16 + fl;                       // h column (GEMM2 K index)
        int hi = (kk >> 4) & 1;
        hLDS[buf][kk >> 5][rg][m + (hi << 4)]
            [kk & 15] = (bf16_t)hv;
      }
    }
    __syncthreads();   // single barrier per f-chunk: all h writers done

    // --- GEMM2: accumulate wave's 128 output columns ---
    #pragma unroll
    for (int kt2 = 0; kt2 < FCHUNK / 32; ++kt2) {
      v16bf a0 = *(const v16bf*)&hLDS[buf][kt2][0][lane][0];
      v16bf a1 = *(const v16bf*)&hLDS[buf][kt2][1][lane][0];
      #pragma unroll
      for (int nt = 0; nt < 8; ++nt) {
        v16bf b = *(const v16bf*)(w2p + (size_t)(nt * 16) * FF + kt2 * 32);
        oacc[0][nt] = __builtin_amdgcn_wmma_f32_16x16x32_bf16(
            false, a0, false, b, (short)0, oacc[0][nt], false, false);
        oacc[1][nt] = __builtin_amdgcn_wmma_f32_16x16x32_bf16(
            false, a1, false, b, (short)0, oacc[1][nt], false, false);
      }
    }
  }

  // --- epilogue: bias2, gate scale, scatter-add into y ---
  #pragma unroll
  for (int nt = 0; nt < 8; ++nt) {
    int d = wave * FCHUNK + nt * 16 + fl;
    float bias2 = b2[e * DM + d];
    #pragma unroll
    for (int rg = 0; rg < NRG; ++rg) {
      #pragma unroll
      for (int j = 0; j < 8; ++j) {
        int m = rg * 16 + j + mrow;
        float g = gLDS[m];
        if (g > 0.f)
          atomic_add_f32(&y[(size_t)tokLDS[m] * DM + d],
                         (oacc[rg][nt][j] + bias2) * g);
      }
    }
  }
}

// ---------------- host launcher ----------------
extern "C" void kernel_launch(void* const* d_in, const int* in_sizes, int n_in,
                              void* d_out, int out_size, void* d_ws, size_t ws_size,
                              hipStream_t stream) {
  const float* x  = (const float*)d_in[0];
  // d_in[1] = route_mask (bool) — unused: route_weight > 0 encodes the mask
  const float* rw = (const float*)d_in[2];
  const float* W1 = (const float*)d_in[3];
  const float* b1 = (const float*)d_in[4];
  const float* W2 = (const float*)d_in[5];
  const float* b2 = (const float*)d_in[6];
  float* y = (float*)d_out;

  char* ws = (char*)d_ws;               // needs ~134.3 MB of workspace
  bf16_t* W1t = (bf16_t*)ws;                 ws += W1T_BYTES;
  bf16_t* W2t = (bf16_t*)ws;                 ws += W2T_BYTES;
  int*    idxb = (int*)ws;                   ws += IDX_BYTES;
  float*  gainb = (float*)ws;                ws += GAIN_BYTES;
  int*    cnt = (int*)ws;

  moe_init <<<1024, 256, 0, stream>>>(y, idxb, gainb, cnt);
  moe_route<<<(T_TOK + 255) / 256, 256, 0, stream>>>(rw, idxb, gainb, cnt);
  moe_prep <<<4096, 256, 0, stream>>>(W1, W2, W1t, W2t);
  moe_ffn  <<<NE * TILES_PER_E, 256, 0, stream>>>(x, b1, b2, W1t, W2t, idxb, gainb, y);
}
